// ConvPMF_78305843741359
// MI455X (gfx1250) — matrix operand order repaired and checked by reading, not compiled, not run on hardware
//
#include <hip/hip_runtime.h>
#include <hip/hip_bf16.h>

typedef unsigned int u32;
typedef __attribute__((ext_vector_type(16))) __bf16 v16bf;
typedef __attribute__((ext_vector_type(8)))  float  v8f;

#define ROWH 226      // halves per x-row in LDS (d 0..225, valid 0..199, pad->224 used)
#define ROWD 113      // dwords per x-row (odd -> conflict-free strided access)
#define WPAD 260      // w rows: w in [-2, 257] -> wp = w+2 in [0,259]
#define FMS  257      // fm / P row stride in floats (odd -> bank-conflict-free column reads)
#define LN2_INV 1.4426950408889634f

__device__ __forceinline__ unsigned short bf16bits(float f) {
    u32 u = __float_as_uint(f);
    u32 r = u + 0x7FFFu + ((u >> 16) & 1u);   // round to nearest even
    return (unsigned short)(r >> 16);
}

// ---------------------------------------------------------------------------
// Kernel 1: zero accumulators + pack conv_w into CDNA5 WMMA A-fragment layout.
// A fragment (16x32 bf16): lane l: m=l&15, half=l>>4; dword v: k0=half*16+v*2
// packed index = (((ft*5+k)*7+ch)*32 + lane)*8 + v   (17920 dwords total)
// ---------------------------------------------------------------------------
__global__ void zero_pack_kernel(const float* __restrict__ conv_w,
                                 float* __restrict__ accum,
                                 u32* __restrict__ apack) {
    int i = blockIdx.x * blockDim.x + threadIdx.x;
    if (i < 4256) accum[i] = 0.0f;                 // item_sum[4096]+factor[32]+dockl[128]
    if (i < 17920) {
        int v    = i & 7;
        int lane = (i >> 3) & 31;
        int rest = i >> 8;
        int ch   = rest % 7;
        int kt   = rest / 7;          // 0..9
        int k    = kt % 5;
        int ft   = kt / 5;
        int m    = lane & 15, half = lane >> 4;
        int k0   = half * 16 + v * 2;
        int d0   = ch * 32 + k0;
        int f    = ft * 16 + m;
        float f0 = (d0     < 200) ? conv_w[(f * 200 + d0    ) * 5 + k] : 0.0f;
        float f1 = (d0 + 1 < 200) ? conv_w[(f * 200 + d0 + 1) * 5 + k] : 0.0f;
        apack[i] = (u32)bf16bits(f0) | ((u32)bf16bits(f1) << 16);
    }
}

// ---------------------------------------------------------------------------
// Kernel 2: one block per review. Gather -> bf16 LDS -> WMMA conv -> tanh fm
// -> softmax stats -> entropy/item atomics -> pairwise-KL partial -> atomic.
// ---------------------------------------------------------------------------
__global__ void __launch_bounds__(256)
convpmf_main(const int* __restrict__ docs, const float* __restrict__ embed,
             const u32* __restrict__ apack,
             float* __restrict__ item_sum, float* __restrict__ factor_sum,
             float* __restrict__ dockl_sum) {
    extern __shared__ __align__(16) char smem[];
    u32*   xd   = (u32*)smem;                               // x: WPAD rows * ROWD dwords
    float* fm   = (float*)(smem + WPAD * ROWH * 2);         // 32 rows * FMS f32 (tanh'd)
    float* P    = fm   + 32 * FMS;                          // 32 rows * FMS f32
    float* mlz  = P    + 32 * FMS;                          // m + logZ per row
    float* ent  = mlz  + 32;                                // natural-log entropy per row
    float* sT   = ent  + 32;                                // block pair-sum

    const int tid  = threadIdx.x;
    const int lane = tid & 31, wave = tid >> 5;
    const int blk  = blockIdx.x;          // review index: b = blk>>4, r = blk&15
    const int b    = blk >> 4;

    if (tid == 0) *sT = 0.0f;

    // zero only the x padding: w rows {0,1,258,259} and d dwords [100,113) everywhere
    for (int i = tid; i < 4 * ROWD; i += 256) {
        int r = i / ROWD, c = i - r * ROWD;
        int wp = (r < 2) ? r : (256 + r);            // 0,1,258,259
        xd[wp * ROWD + c] = 0u;
    }
    for (int i = tid; i < WPAD * 13; i += 256) {
        int wp = i / 13, c = 100 + (i - wp * 13);
        xd[wp * ROWD + c] = 0u;
    }
    __syncthreads();

    // gather: 4 threads per word, each converts 25 float2 -> 25 packed-bf16 dwords
    for (int wbase = 0; wbase < 256; wbase += 64) {
        int w   = wbase + (tid >> 2);
        int tok = docs[blk * 256 + w];
        const float2* row = (const float2*)(embed + (size_t)tok * 200);
        int sub = tid & 3;                 // dword chunk [sub*25, sub*25+25)
        u32* dst = xd + (w + 2) * ROWD + sub * 25;
        #pragma unroll
        for (int j = 0; j < 25; ++j) {
            float2 f2 = row[sub * 25 + j];
            union { __bf16 h[2]; u32 u; } pk;
            pk.h[0] = (__bf16)f2.x; pk.h[1] = (__bf16)f2.y;
            dst[j] = pk.u;
        }
    }
    __syncthreads();

    // conv as GEMM: fm[f,w] = sum_{k,d} conv_w[f,d,k] * x[d, w+k-2]
    {
        const int ft  = wave & 1;           // f tile (0,1)
        const int wt0 = (wave >> 1) * 4;    // 4 w-tiles per wave
        const int n = lane & 15, half = lane >> 4;
        v8f c[4];
        #pragma unroll
        for (int i = 0; i < 4; ++i)
            #pragma unroll
            for (int q = 0; q < 8; ++q) c[i][q] = 0.0f;

        for (int k = 0; k < 5; ++k) {
            for (int ch = 0; ch < 7; ++ch) {
                // A fragment: contiguous 32B per lane from L2-hot packed weights
                v16bf a = *(const v16bf*)(apack + ((((ft * 5 + k) * 7 + ch) * 32 + lane) << 3));
                // stage all 4 B fragments first so ds_loads overlap the wmmas
                union { u32 u[8]; v16bf v; } bu[4];
                #pragma unroll
                for (int i = 0; i < 4; ++i) {
                    // B fragment (32x16): K = ch*32 + half*16 + 2v, N = n, w = wt*16+n+k-2
                    int wp = (wt0 + i) * 16 + n + k;          // wp = w + 2 + (k-2)
                    const u32* rp = xd + wp * ROWD + ch * 16 + half * 8;
                    #pragma unroll
                    for (int q = 0; q < 8; ++q) bu[i].u[q] = rp[q];
                }
                #pragma unroll
                for (int i = 0; i < 4; ++i)
                    c[i] = __builtin_amdgcn_wmma_f32_16x16x32_bf16(
                        false, a, false, bu[i].v, (short)0, c[i], false, false);
            }
        }
        #pragma unroll
        for (int i = 0; i < 4; ++i) {
            int wcol = (wt0 + i) * 16 + n;
            #pragma unroll
            for (int v = 0; v < 8; ++v) {
                int f = ft * 16 + half * 8 + v;   // C/D layout: M = v + half*8
                fm[f * FMS + wcol] = tanhf(c[i][v]);
            }
        }
    }
    __syncthreads();

    // per-row (f) softmax stats over W=256; rows f = wave*4 + j
    for (int j = 0; j < 4; ++j) {
        int f = wave * 4 + j;
        float v[8];
        #pragma unroll
        for (int t = 0; t < 8; ++t) v[t] = fm[f * FMS + lane + 32 * t];
        float m = v[0];
        #pragma unroll
        for (int t = 1; t < 8; ++t) m = fmaxf(m, v[t]);
        for (int off = 16; off > 0; off >>= 1) m = fmaxf(m, __shfl_xor(m, off, 32));
        float Z = 0.0f, S1 = 0.0f;
        #pragma unroll
        for (int t = 0; t < 8; ++t) { float x = v[t] - m, e = __expf(x); Z += e; S1 += x * e; }
        for (int off = 16; off > 0; off >>= 1) { Z += __shfl_xor(Z, off, 32); S1 += __shfl_xor(S1, off, 32); }
        float logZ = __logf(Z);
        float entn = logZ - S1 / Z;          // -sum p*logp (natural)
        float rZ   = 1.0f / Z;
        #pragma unroll
        for (int t = 0; t < 8; ++t) P[f * FMS + lane + 32 * t] = __expf(v[t] - m) * rZ;
        if (lane == 0) {
            mlz[f] = m + logZ; ent[f] = entn;
            atomicAdd(&item_sum[b * 32 + f], m);                 // max over W of tanh fm
            atomicAdd(&factor_sum[f], entn * LN2_INV);           // entropy in bits
        }
    }
    __syncthreads();

    // pairwise KL: sum_{f<g} (self[f] - cross[f,g]),
    // cross[f,g] = (sum_w P[f,w]*fm[g,w]) - (m_g + logZ_g),  self[f] = -ent[f]
    float local = 0.0f;
    for (int p = tid; p < 496; p += 256) {
        int f = 0, rem = p, cnt = 31;
        while (rem >= cnt) { rem -= cnt; ++f; --cnt; }
        int g = f + 1 + rem;
        const float* Pf = P  + f * FMS;
        const float* Fg = fm + g * FMS;
        float dot = 0.0f;
        for (int w = 0; w < 256; ++w) dot += Pf[w] * Fg[w];
        local += (-ent[f]) - (dot - mlz[g]);
    }
    atomicAdd(sT, local);
    __syncthreads();
    if (tid == 0) atomicAdd(&dockl_sum[b], *sT);
}

// ---------------------------------------------------------------------------
// Kernel 3: finalize the 162 outputs.
// ---------------------------------------------------------------------------
__global__ void finalize_kernel(const int* __restrict__ user_idx,
                                const float* __restrict__ w_user,
                                const float* __restrict__ bias,
                                const float* __restrict__ item_sum,
                                const float* __restrict__ factor_sum,
                                const float* __restrict__ dockl_sum,
                                float* __restrict__ out) {
    int t = threadIdx.x;
    if (t < 128) {
        const float* ue = w_user + (size_t)user_idx[t] * 32;
        const float* it = item_sum + t * 32;
        float s = 0.0f;
        for (int f = 0; f < 32; ++f) s += ue[f] * (it[f] * (1.0f / 16.0f));
        out[t] = s + bias[0];
    }
    if (t < 32) out[129 + t] = factor_sum[t] * (1.0f / 2048.0f);    // /(B*R)
    if (t == 0) {
        float s = 0.0f;
        for (int f = 0; f < 32; ++f) s += factor_sum[f];
        out[128] = s / (2048.0f * 32.0f);                           // /(B*R*F)
        float kd = 0.0f;
        for (int bb = 0; bb < 128; ++bb) kd += dockl_sum[bb];
        out[161] = kd / (16.0f * 496.0f * 128.0f);                  // /R /n_pairs /B
    }
}

extern "C" void kernel_launch(void* const* d_in, const int* in_sizes, int n_in,
                              void* d_out, int out_size, void* d_ws, size_t ws_size,
                              hipStream_t stream) {
    const int*   user_idx = (const int*)d_in[0];
    const int*   docs     = (const int*)d_in[1];
    const float* w_user   = (const float*)d_in[2];
    const float* embed    = (const float*)d_in[3];
    const float* conv_w   = (const float*)d_in[4];
    const float* bias     = (const float*)d_in[5];
    float* out = (float*)d_out;

    float* accum      = (float*)d_ws;
    float* item_sum   = accum;            // 128*32
    float* factor_sum = accum + 4096;     // 32
    float* dockl_sum  = accum + 4128;     // 128
    u32*   apack      = (u32*)((char*)d_ws + 32768);   // 17920 dwords

    zero_pack_kernel<<<70, 256, 0, stream>>>(conv_w, accum, apack);

    size_t lds = (size_t)WPAD * ROWH * 2      // x (bf16)
               + (size_t)32 * FMS * 4 * 2     // fm + P (f32, stride-257)
               + (32 * 2 + 1) * 4 + 16;       // row scalars + pair sum
    convpmf_main<<<2048, 256, lds, stream>>>(docs, embed, apack,
                                             item_sum, factor_sum, dockl_sum);

    finalize_kernel<<<1, 128, 0, stream>>>(user_idx, w_user, bias,
                                           item_sum, factor_sum, dockl_sum, out);
}
